// RuleBasedEngine_7344394076241
// MI455X (gfx1250) — matrix operand order repaired and checked by reading, not compiled, not run on hardware
//
#include <hip/hip_runtime.h>

typedef __attribute__((ext_vector_type(16))) _Float16 v16h;
typedef __attribute__((ext_vector_type(8)))  _Float16 v8h;
typedef __attribute__((ext_vector_type(4)))  _Float16 v4h;
typedef __attribute__((ext_vector_type(8)))  float    v8f;

#define D_MODEL   512
#define NUM_RULES 16
#define BATCH     4
#define SEQ       4096
#define ROWS      (BATCH * SEQ)      // 16384
#define TILE_M    64
#define PITCH     520                // halves per LDS row (512 + 8 pad -> bank-conflict free)
#define WELEMS    (D_MODEL * D_MODEL)
#define KBLOCKS   (D_MODEL / 32)     // 16
#define FRAGH     (32 * 16)          // halves per fragment (1 KB)
#define NTILEH    (KBLOCKS * FRAGH)  // halves per n-tile strip (16 KB)

static __device__ inline v8f wmma16(v16h a, v16h b, v8f c) {
  return __builtin_amdgcn_wmma_f32_16x16x32_f16(false, a, false, b, (short)0, c,
                                                false, false);
}

// A-fragment (16x32 f16) from an LDS tile, row-major with PITCH halves/row.
// lanes 0-15: row M=lane, K = kb+0..7 and kb+16..23; lanes 16-31: K = kb+8..15, kb+24..31.
static __device__ inline v16h load_a_lds(const _Float16* base, int mrow, int kb, int lane) {
  int row  = mrow + (lane & 15);
  int koff = kb + ((lane < 16) ? 0 : 8);
  const _Float16* p = base + row * PITCH + koff;
  v8h lo = *(const v8h*)p;
  v8h hi = *(const v8h*)(p + 16);
  v16h r;
#pragma unroll
  for (int i = 0; i < 8; ++i) { r[i] = lo[i]; r[8 + i] = hi[i]; }
  return r;
}

// B-fragment from fragment-major packed weights: lane's 32 bytes are contiguous at p.
static __device__ inline v16h load_frag(const _Float16* __restrict__ p) {
  v8h lo = *(const v8h*)p;
  v8h hi = *(const v8h*)(p + 8);
  v16h r;
#pragma unroll
  for (int i = 0; i < 8; ++i) { r[i] = lo[i]; r[8 + i] = hi[i]; }
  return r;
}

static __device__ inline float sigmoidf_fast(float v) {
  return 1.f / (1.f + __expf(-v));
}

#define ZERO8F (v8f){0.f,0.f,0.f,0.f,0.f,0.f,0.f,0.f}

// Dual n-tile, 4 m-tile GEMM panel, ping-pong B prefetch.
// acc[0..3]: n0 tiles m=0..3 ; acc[4..7]: n1=n0+16 tiles m=0..3.
// B from fragment-major layout: frag(ntile,kblock) at ((ntile*16+kblock)*32+lane)*16 halves.
static __device__ inline void gemm_panel2(const _Float16* lds, const _Float16* __restrict__ wt,
                                          int n0, int lane, v8f acc[8]) {
  const _Float16* p0 = wt + ((size_t)(n0 >> 4) * NTILEH) + lane * 16;  // ntile0, kblock0
  const _Float16* p1 = p0 + NTILEH;                                    // ntile1
  v16h b00 = load_frag(p0);
  v16h b10 = load_frag(p1);
  v16h b01 = load_frag(p0 + 512);
  v16h b11 = load_frag(p1 + 512);
#pragma unroll 1
  for (int kk = 0; kk < KBLOCKS - 2; kk += 2) {
    int kb = kk * 32;
    {
      v16h a0 = load_a_lds(lds,  0, kb, lane);
      v16h a1 = load_a_lds(lds, 16, kb, lane);
      v16h a2 = load_a_lds(lds, 32, kb, lane);
      v16h a3 = load_a_lds(lds, 48, kb, lane);
      acc[0] = wmma16(a0, b00, acc[0]);
      acc[1] = wmma16(a1, b00, acc[1]);
      acc[2] = wmma16(a2, b00, acc[2]);
      acc[3] = wmma16(a3, b00, acc[3]);
      acc[4] = wmma16(a0, b10, acc[4]);
      acc[5] = wmma16(a1, b10, acc[5]);
      acc[6] = wmma16(a2, b10, acc[6]);
      acc[7] = wmma16(a3, b10, acc[7]);
      b00 = load_frag(p0 + (kk + 2) * 512);
      b10 = load_frag(p1 + (kk + 2) * 512);
    }
    {
      v16h a0 = load_a_lds(lds,  0, kb + 32, lane);
      v16h a1 = load_a_lds(lds, 16, kb + 32, lane);
      v16h a2 = load_a_lds(lds, 32, kb + 32, lane);
      v16h a3 = load_a_lds(lds, 48, kb + 32, lane);
      acc[0] = wmma16(a0, b01, acc[0]);
      acc[1] = wmma16(a1, b01, acc[1]);
      acc[2] = wmma16(a2, b01, acc[2]);
      acc[3] = wmma16(a3, b01, acc[3]);
      acc[4] = wmma16(a0, b11, acc[4]);
      acc[5] = wmma16(a1, b11, acc[5]);
      acc[6] = wmma16(a2, b11, acc[6]);
      acc[7] = wmma16(a3, b11, acc[7]);
      b01 = load_frag(p0 + (kk + 3) * 512);
      b11 = load_frag(p1 + (kk + 3) * 512);
    }
  }
  {  // tail: kb = 448 (b00/b10), kb = 480 (b01/b11)
    const int kb = D_MODEL - 64;
    v16h a0 = load_a_lds(lds,  0, kb, lane);
    v16h a1 = load_a_lds(lds, 16, kb, lane);
    v16h a2 = load_a_lds(lds, 32, kb, lane);
    v16h a3 = load_a_lds(lds, 48, kb, lane);
    acc[0] = wmma16(a0, b00, acc[0]);
    acc[1] = wmma16(a1, b00, acc[1]);
    acc[2] = wmma16(a2, b00, acc[2]);
    acc[3] = wmma16(a3, b00, acc[3]);
    acc[4] = wmma16(a0, b10, acc[4]);
    acc[5] = wmma16(a1, b10, acc[5]);
    acc[6] = wmma16(a2, b10, acc[6]);
    acc[7] = wmma16(a3, b10, acc[7]);
    a0 = load_a_lds(lds,  0, kb + 32, lane);
    a1 = load_a_lds(lds, 16, kb + 32, lane);
    a2 = load_a_lds(lds, 32, kb + 32, lane);
    a3 = load_a_lds(lds, 48, kb + 32, lane);
    acc[0] = wmma16(a0, b01, acc[0]);
    acc[1] = wmma16(a1, b01, acc[1]);
    acc[2] = wmma16(a2, b01, acc[2]);
    acc[3] = wmma16(a3, b01, acc[3]);
    acc[4] = wmma16(a0, b11, acc[4]);
    acc[5] = wmma16(a1, b11, acc[5]);
    acc[6] = wmma16(a2, b11, acc[6]);
    acc[7] = wmma16(a3, b11, acc[7]);
  }
}

// ---------------------------------------------------------------- convert x
__global__ __launch_bounds__(256) void k_convert_x(const float* __restrict__ x,
                                                   _Float16* __restrict__ xh) {
  size_t c = (size_t)blockIdx.x * blockDim.x + threadIdx.x;  // 16B chunk id
  const float4* p = (const float4*)(x + c * 8);
  float4 f0 = p[0], f1 = p[1];
  v8h h;
  h[0] = (_Float16)f0.x; h[1] = (_Float16)f0.y; h[2] = (_Float16)f0.z; h[3] = (_Float16)f0.w;
  h[4] = (_Float16)f1.x; h[5] = (_Float16)f1.y; h[6] = (_Float16)f1.z; h[7] = (_Float16)f1.w;
  *(v8h*)(xh + c * 8) = h;
}

// ---------------- pack weights W[k][n] (f32) -> fragment-major f16
// frag(ntile,kblock): 1 KB block; lane l: col = ntile*16+(l&15),
// halves h=0..15 -> kin = (h<8 ? 0 : 8) + (l<16 ? 0 : 8) + h.
__global__ __launch_bounds__(256) void k_pack_w(const float* __restrict__ Wc,
                                                const float* __restrict__ Wa1,
                                                const float* __restrict__ Wa2,
                                                _Float16* __restrict__ Wcp,
                                                _Float16* __restrict__ W1p,
                                                _Float16* __restrict__ W2p) {
  __shared__ _Float16 tile[32][33];  // [kin][nloc]
  int z = blockIdx.z;
  int which = z >> 4, r = z & 15;
  const float* W = (which == 0) ? Wc : (which == 1) ? Wa1 : Wa2;
  _Float16* Wp   = (which == 0) ? Wcp : (which == 1) ? W1p : W2p;
  W  += (size_t)r * WELEMS;
  Wp += (size_t)r * WELEMS;
  int nb = blockIdx.x * 32;  // n base (2 n-tiles)
  int kb = blockIdx.y * 32;  // k base (1 k-block)
  int tx = threadIdx.x, ty = threadIdx.y;  // 32 x 8
#pragma unroll
  for (int j = 0; j < 32; j += 8)
    tile[ty + j][tx] = (_Float16)W[(size_t)(kb + ty + j) * D_MODEL + nb + tx];
  __syncthreads();
  int c = ty * 32 + tx;          // 0..255 : chunk of 4 halves
  int subtile = c >> 7;          // 0..1
  int fl = (c >> 2) & 31;        // fragment lane 0..31
  int q = c & 3;                 // half-group 0..3 (halves q*4..q*4+3)
  int nloc = subtile * 16 + (fl & 15);
  int kin0 = ((q < 2) ? 0 : 8) + ((fl < 16) ? 0 : 8) + q * 4;
  v4h v;
#pragma unroll
  for (int t = 0; t < 4; ++t) v[t] = tile[kin0 + t][nloc];
  int ntile = blockIdx.x * 2 + subtile;
  *(v4h*)(Wp + ((size_t)ntile * KBLOCKS + blockIdx.y) * FRAGH + fl * 16 + q * 4) = v;
}

// ------------------------------------------------- cond partial sums
// grid: 256 = r(16) x b(4) x chunk(4); block: 512 (16 waves, each owns 32 cols)
__global__ __launch_bounds__(512) void k_cond(const _Float16* __restrict__ xh,
                                              const _Float16* __restrict__ Wcp,
                                              const float* __restrict__ bc,
                                              float* __restrict__ partial) {
  __shared__ _Float16 xs[TILE_M * PITCH];
  __shared__ float condacc[D_MODEL];
  int tid = threadIdx.x, lane = tid & 31, w = tid >> 5;
  int bx = blockIdx.x;
  int r = bx >> 4, b = (bx >> 2) & 3, chunk = bx & 3;
  const _Float16* wc = Wcp + (size_t)r * WELEMS;
  condacc[tid] = 0.f;
  __syncthreads();
  int nb_w = w * 32, col15 = lane & 15;
#pragma unroll 1
  for (int tile = 0; tile < 16; ++tile) {
    int row0 = b * SEQ + chunk * 1024 + tile * TILE_M;
#pragma unroll
    for (int c = tid; c < TILE_M * 64; c += 512) {
      int rr = c >> 6, cc = c & 63;
      *(v8h*)&xs[rr * PITCH + cc * 8] =
          *(const v8h*)(xh + (size_t)(row0 + rr) * D_MODEL + cc * 8);
    }
    __syncthreads();
    {
      v8f acc[8];
#pragma unroll
      for (int m = 0; m < 8; ++m) acc[m] = ZERO8F;
      gemm_panel2(xs, wc, nb_w, lane, acc);
#pragma unroll
      for (int j = 0; j < 2; ++j) {
        int n = nb_w + j * 16;
        float bcv = bc[n + col15];
        float ps = 0.f;
#pragma unroll
        for (int m = 0; m < 4; ++m)
#pragma unroll
          for (int i = 0; i < 8; ++i) ps += sigmoidf_fast(acc[j * 4 + m][i] + bcv);
        ps += __shfl_xor(ps, 16);          // combine rows 0-7 with rows 8-15
        if (lane < 16) condacc[n + lane] += ps;  // wave-private columns: no race
      }
    }
    __syncthreads();
  }
  partial[(size_t)bx * D_MODEL + tid] = condacc[tid];
}

// ------------------------------------------------- cond reduce: mean over 4 chunks
__global__ __launch_bounds__(256) void k_reduce(const float* __restrict__ partial,
                                                float* __restrict__ cond) {
  int idx = blockIdx.x * 256 + threadIdx.x;  // 0..16*4*512-1
  int rb = idx >> 9, col = idx & 511;
  float s = 0.f;
#pragma unroll
  for (int c = 0; c < 4; ++c) s += partial[(size_t)((rb << 2) + c) * D_MODEL + col];
  cond[idx] = s * (1.f / (float)SEQ);
}

// ------------------------------------------------- main: gated MLP, sum over rules
// grid: 256 row-tiles of 64; block: 512 (16 waves); wave w owns cols [w*32, w*32+32)
__global__ __launch_bounds__(512) void k_main(const _Float16* __restrict__ xh,
                                              const _Float16* __restrict__ W1p,
                                              const _Float16* __restrict__ W2p,
                                              const float* __restrict__ ba1,
                                              const float* __restrict__ ba2,
                                              const float* __restrict__ cond,
                                              float* __restrict__ out) {
  __shared__ _Float16 xs[TILE_M * PITCH];
  __shared__ _Float16 hs[TILE_M * PITCH];
  int tid = threadIdx.x, lane = tid & 31, w = tid >> 5;
  int row0 = blockIdx.x * TILE_M;
  int b = row0 >> 12;  // 4096 rows per batch
  // stage X tile
#pragma unroll
  for (int c = tid; c < TILE_M * 64; c += 512) {
    int rr = c >> 6, cc = c & 63;
    *(v8h*)&xs[rr * PITCH + cc * 8] =
        *(const v8h*)(xh + (size_t)(row0 + rr) * D_MODEL + cc * 8);
  }
  __syncthreads();

  int nb_w = w * 32, col15 = lane & 15;
  int rhalf = (lane < 16) ? 0 : 8;

  v8f outacc[8];
#pragma unroll
  for (int i = 0; i < 8; ++i) outacc[i] = ZERO8F;

#pragma unroll 1
  for (int r = 0; r < NUM_RULES; ++r) {
    const _Float16* w1 = W1p + (size_t)r * WELEMS;
    const _Float16* w2 = W2p + (size_t)r * WELEMS;
    const float* condr = cond + (size_t)(r * 4 + b) * D_MODEL;

    // ---- GEMM1: H = silu(X @ Wa1 + ba1) -> LDS
    {
      v8f acc[8];
#pragma unroll
      for (int m = 0; m < 8; ++m) acc[m] = ZERO8F;
      gemm_panel2(xs, w1, nb_w, lane, acc);
#pragma unroll
      for (int j = 0; j < 2; ++j) {
        int n = nb_w + j * 16;
        float b1v = ba1[n + col15];
#pragma unroll
        for (int m = 0; m < 4; ++m) {
          int rowb = m * 16 + rhalf;
#pragma unroll
          for (int i = 0; i < 8; ++i) {
            float v = acc[j * 4 + m][i] + b1v;
            float s = v * sigmoidf_fast(v);
            hs[(rowb + i) * PITCH + n + col15] = (_Float16)s;
          }
        }
      }
    }
    __syncthreads();

    // ---- GEMM2: out += (H @ Wa2 + ba2) * cond
    {
      v8f acc[8];
#pragma unroll
      for (int m = 0; m < 8; ++m) acc[m] = ZERO8F;
      gemm_panel2(hs, w2, nb_w, lane, acc);
#pragma unroll
      for (int j = 0; j < 2; ++j) {
        int n = nb_w + j * 16;
        float cv  = condr[n + col15];
        float b2v = ba2[n + col15];
#pragma unroll
        for (int m = 0; m < 4; ++m)
#pragma unroll
          for (int i = 0; i < 8; ++i)
            outacc[j * 4 + m][i] += (acc[j * 4 + m][i] + b2v) * cv;
      }
    }
    __syncthreads();  // hs reused next rule
  }

  // ---- store
#pragma unroll
  for (int j = 0; j < 2; ++j)
#pragma unroll
    for (int m = 0; m < 4; ++m)
#pragma unroll
      for (int i = 0; i < 8; ++i)
        out[(size_t)(row0 + m * 16 + rhalf + i) * D_MODEL + nb_w + j * 16 + col15] =
            outacc[j * 4 + m][i];
}

extern "C" void kernel_launch(void* const* d_in, const int* in_sizes, int n_in,
                              void* d_out, int out_size, void* d_ws, size_t ws_size,
                              hipStream_t stream) {
  const float* x   = (const float*)d_in[0];
  const float* Wc  = (const float*)d_in[1];
  const float* bc  = (const float*)d_in[2];
  const float* Wa1 = (const float*)d_in[3];
  const float* ba1 = (const float*)d_in[4];
  const float* Wa2 = (const float*)d_in[5];
  const float* ba2 = (const float*)d_in[6];
  float* out = (float*)d_out;

  char* ws = (char*)d_ws;
  _Float16* xh   = (_Float16*)(ws);                       // 16384*512*2   = 16,777,216
  _Float16* Wcp  = (_Float16*)(ws + 16777216);            // 16*512*512*2  =  8,388,608
  _Float16* W1p  = (_Float16*)(ws + 25165824);
  _Float16* W2p  = (_Float16*)(ws + 33554432);
  float*    cond = (float*)   (ws + 41943040);            // 16*4*512*4    =    131,072
  float*    part = (float*)   (ws + 42074112);            // 16*4*4*512*4  =    524,288

  k_convert_x<<<ROWS * D_MODEL / (256 * 8), 256, 0, stream>>>(x, xh);
  k_pack_w<<<dim3(16, 16, 48), dim3(32, 8), 0, stream>>>(Wc, Wa1, Wa2, Wcp, W1p, W2p);
  k_cond<<<256, 512, 0, stream>>>(xh, Wcp, bc, part);
  k_reduce<<<NUM_RULES * BATCH * D_MODEL / 256, 256, 0, stream>>>(part, cond);
  k_main<<<ROWS / TILE_M, 512, 0, stream>>>(xh, W1p, W2p, ba1, ba2, cond, out);
}
